// SmallLanguageModel_62156766707966
// MI455X (gfx1250) — matrix operand order repaired and checked by reading, not compiled, not run on hardware
//
#include <hip/hip_runtime.h>
#include <hip/hip_bf16.h>
#include <math.h>

typedef __attribute__((ext_vector_type(16))) __bf16 v16bf;
typedef __attribute__((ext_vector_type(8)))  float  v8f;
typedef unsigned short us16;

#define LCOUNT 2
#define BBATCH 4
#define TSEQ   512
#define DMODEL 768
#define NHEAD  12
#define HSZ    64
#define FFDIM  3072
#define VOCAB  50257

__device__ __forceinline__ us16 f2bf(float f) {
    unsigned u = __float_as_uint(f);
    u += 0x7FFFu + ((u >> 16) & 1u);   // round-to-nearest-even
    return (us16)(u >> 16);
}

union FragU { v16bf v; uint4 q[2]; };

enum { GF_GELU = 1, GF_ACC = 2, GF_CBF16 = 4, GF_STORET = 8 };

// CDNA5 async global->LDS copy (ASYNCcnt path). Per-lane LDS dest address.
__device__ __forceinline__ void async_cp16(unsigned ldsOff, const void* g) {
    asm volatile("global_load_async_to_lds_b128 %0, %1, off"
                 :: "v"(ldsOff), "v"(g) : "memory");
}
__device__ __forceinline__ void wait_async0() {
    asm volatile("s_wait_asynccnt 0x0" ::: "memory");
}

// ---------------------------------------------------------------------------
// Batched bf16 WMMA GEMM: C = A[MxK] x Bt[NxK]^T (+bias)(+gelu)(+acc)
// A, Bt bf16; B ALWAYS stored [N][K] so both LDS tiles are contiguous along K
// and staged with global_load_async_to_lds_b128, double-buffered so tile k+1
// streams in (ASYNCcnt) while tile k runs through the WMMA pipe.
// Block tile 128x128, K-chunk 32; 8 wave32, each wave 32x64 (2x4 fragments).
// Requires: M % 128 == 0 and gridDim.y == M/128 (true for all call sites).
// Batch offset = (z/divi)*outer + (z%divi)*inner for each of A/B/C.
// ---------------------------------------------------------------------------
template <int FLAGS>
__global__ __launch_bounds__(256) void gemm_bf16_wmma(
    const us16* __restrict__ A, long long lda, long long sAo, long long sAi,
    const us16* __restrict__ Bt, long long ldb, long long sBo, long long sBi,
    void* __restrict__ Cv, long long ldc, long long sCo, long long sCi,
    const float* __restrict__ bias, long long sBiasI,
    int M, int N, int K, int divi)
{
    constexpr int TILE = 128 * 32;            // bf16 elements per tile
    __shared__ __align__(16) us16 As[2 * TILE];
    __shared__ __align__(16) us16 Bs[2 * TILE];

    const int batch = blockIdx.z;
    const int bo = batch / divi, bi = batch % divi;
    A  += bo * sAo + bi * sAi;
    Bt += bo * sBo + bi * sBi;
    const long long cBase = bo * sCo + bi * sCi;
    if (bias) bias += bi * sBiasI;

    const int rowBase = blockIdx.y * 128;
    const int colBase = blockIdx.x * 128;
    const int tid  = threadIdx.x;
    const int lane = tid & 31;
    const int wid  = tid >> 5;
    const int wm   = (wid & 3) * 32;
    const int wn   = (wid >> 2) * 64;
    const int laneM = lane & 15, laneH = lane >> 4;

    const unsigned ldsA = (unsigned)(size_t)&As[0];
    const unsigned ldsB = (unsigned)(size_t)&Bs[0];

    // staging map: thread -> (tile row, 16-element half-row); 4 asyncs/thread
    const int sr = tid >> 1;            // 0..127
    const int sk = (tid & 1) * 16;      // 0 or 16 (bf16 elements)
    int gnS = colBase + sr; if (gnS >= N) gnS = N - 1;      // clamp (no branch)
    const us16* gArow = A  + (long long)(rowBase + sr) * lda + sk;
    const us16* gBrow = Bt + (long long)gnS * ldb + sk;
    const unsigned lOff = (unsigned)((sr * 32 + sk) * 2);

    auto stage = [&](int kblk, int buf) {
        const long long kk = (long long)kblk << 5;
        const us16* ga = gArow + kk;
        const us16* gb = gBrow + kk;
        unsigned la = ldsA + (unsigned)(buf * TILE * 2) + lOff;
        unsigned lb = ldsB + (unsigned)(buf * TILE * 2) + lOff;
        async_cp16(la,      ga);
        async_cp16(la + 16, ga + 8);
        async_cp16(lb,      gb);
        async_cp16(lb + 16, gb + 8);
    };

    v8f acc[2][4] = {};
    const int nk = K >> 5;

    stage(0, 0);
    for (int kb = 0; kb < nk; ++kb) {
        const int cur = kb & 1;
        wait_async0();          // only tile-kb copies are outstanding here
        __syncthreads();        // tile kb visible; prior reads of buf 1-cur done
        if (kb + 1 < nk) stage(kb + 1, 1 - cur);   // overlaps with WMMAs below

        const us16* Ab = &As[cur * TILE];
        const us16* Bb = &Bs[cur * TILE];

        FragU ua, ub;
        v16bf afr[2], bfr[4];
        #pragma unroll
        for (int i = 0; i < 2; ++i) {
            int row = wm + i * 16 + laneM;
            ua.q[0] = *reinterpret_cast<const uint4*>(&Ab[row * 32 + laneH * 8]);
            ua.q[1] = *reinterpret_cast<const uint4*>(&Ab[row * 32 + 16 + laneH * 8]);
            afr[i] = ua.v;
        }
        #pragma unroll
        for (int j = 0; j < 4; ++j) {
            int col = wn + j * 16 + laneM;
            ub.q[0] = *reinterpret_cast<const uint4*>(&Bb[col * 32 + laneH * 8]);
            ub.q[1] = *reinterpret_cast<const uint4*>(&Bb[col * 32 + 16 + laneH * 8]);
            bfr[j] = ub.v;
        }
        #pragma unroll
        for (int i = 0; i < 2; ++i)
            #pragma unroll
            for (int j = 0; j < 4; ++j)
                acc[i][j] = __builtin_amdgcn_wmma_f32_16x16x32_bf16(
                    false, afr[i], false, bfr[j], (short)0, acc[i][j], false, false);
    }

    // ---- epilogue (specialized at compile time via FLAGS) ----
    float* Cf = (float*)Cv;
    us16*  Ch = (us16*)Cv;
    #pragma unroll
    for (int i = 0; i < 2; ++i) {
        #pragma unroll
        for (int j = 0; j < 4; ++j) {
            int gc = colBase + wn + j * 16 + laneM;
            if (gc >= N) continue;
            float bvv = bias ? bias[gc] : 0.0f;
            #pragma unroll
            for (int r = 0; r < 8; ++r) {
                int gr = rowBase + wm + i * 16 + r + laneH * 8;
                float v = acc[i][j][r] + bvv;
                if (FLAGS & GF_GELU) v = 0.5f * v * (1.0f + erff(v * 0.70710678118654752f));
                long long idx = (FLAGS & GF_STORET) ? ((long long)gc * ldc + gr)
                                                    : ((long long)gr * ldc + gc);
                idx += cBase;
                if (FLAGS & GF_ACC)        Cf[idx] += v;
                else if (FLAGS & GF_CBF16) Ch[idx] = f2bf(v);
                else                       Cf[idx] = v;
            }
        }
    }
}

// ---------------------------------------------------------------------------
// fp32 -> bf16 elementwise (vector of 4)
// ---------------------------------------------------------------------------
__global__ __launch_bounds__(256) void conv_bf16_kernel(
    const float* __restrict__ in, us16* __restrict__ out, long long n4)
{
    long long i = (long long)blockIdx.x * 256 + threadIdx.x;
    if (i >= n4) return;
    float4 v = *reinterpret_cast<const float4*>(in + i * 4);
    us16* o = out + i * 4;
    o[0] = f2bf(v.x); o[1] = f2bf(v.y); o[2] = f2bf(v.z); o[3] = f2bf(v.w);
}

// ---------------------------------------------------------------------------
// Tiled transpose+convert: in[z][R][C] fp32 -> out[z][C][R] bf16
// ---------------------------------------------------------------------------
__global__ void transpose_conv_kernel(
    const float* __restrict__ in, us16* __restrict__ out, int R, int C)
{
    __shared__ float tile[32][33];
    const float* ip = in + (long long)blockIdx.z * R * C;
    us16* op = out + (long long)blockIdx.z * R * C;
    int r0 = blockIdx.y * 32, c0 = blockIdx.x * 32;
    for (int i = threadIdx.y; i < 32; i += 8) {
        int r = r0 + i, c = c0 + threadIdx.x;
        tile[i][threadIdx.x] = (r < R && c < C) ? ip[(long long)r * C + c] : 0.f;
    }
    __syncthreads();
    for (int i = threadIdx.y; i < 32; i += 8) {
        int c = c0 + i, r = r0 + threadIdx.x;
        if (c < C && r < R) op[(long long)c * R + r] = f2bf(tile[threadIdx.x][i]);
    }
}

// ---------------------------------------------------------------------------
// x[b,t,d] = wte[idx[b,t], d] + wpe[t, d]   (fp32 residual stream)
// ---------------------------------------------------------------------------
__global__ __launch_bounds__(256) void embed_kernel(
    const int* __restrict__ idx, const float* __restrict__ wte,
    const float* __restrict__ wpe, float* __restrict__ x)
{
    long long i = (long long)blockIdx.x * 256 + threadIdx.x;
    int d = (int)(i % DMODEL);
    long long bt = i / DMODEL;
    int t = (int)(bt % TSEQ);
    x[i] = wte[(long long)idx[bt] * DMODEL + d] + wpe[(long long)t * DMODEL + d];
}

// ---------------------------------------------------------------------------
// Row LayerNorm over D=768, fp32 in -> bf16 out; one block per row
// ---------------------------------------------------------------------------
__global__ __launch_bounds__(256) void ln_kernel(
    const float* __restrict__ x, const float* __restrict__ s,
    const float* __restrict__ b, us16* __restrict__ out)
{
    __shared__ float red[256];
    const float* xr = x + (long long)blockIdx.x * DMODEL;
    float sum = 0.f;
    for (int i = threadIdx.x; i < DMODEL; i += 256) sum += xr[i];
    red[threadIdx.x] = sum; __syncthreads();
    for (int st = 128; st > 0; st >>= 1) {
        if (threadIdx.x < st) red[threadIdx.x] += red[threadIdx.x + st];
        __syncthreads();
    }
    float mean = red[0] / DMODEL;
    __syncthreads();
    float vs = 0.f;
    for (int i = threadIdx.x; i < DMODEL; i += 256) { float d = xr[i] - mean; vs += d * d; }
    red[threadIdx.x] = vs; __syncthreads();
    for (int st = 128; st > 0; st >>= 1) {
        if (threadIdx.x < st) red[threadIdx.x] += red[threadIdx.x + st];
        __syncthreads();
    }
    float rstd = rsqrtf(red[0] / DMODEL + 1e-5f);
    us16* o = out + (long long)blockIdx.x * DMODEL;
    for (int i = threadIdx.x; i < DMODEL; i += 256)
        o[i] = f2bf((xr[i] - mean) * rstd * s[i] + b[i]);
}

// ---------------------------------------------------------------------------
// Causal softmax of one fp32 score row -> bf16 prob row, * 1/sqrt(HS)
// ---------------------------------------------------------------------------
__global__ __launch_bounds__(256) void attn_softmax_kernel(
    const float* __restrict__ scores, us16* __restrict__ probs)
{
    __shared__ float red[256];
    const int row = blockIdx.x;          // b*H*T + h*T + t
    const int t = row % TSEQ;
    const int valid = t + 1;
    const float* p = scores + (long long)row * TSEQ;
    us16* o = probs + (long long)row * TSEQ;

    float m = -INFINITY;
    for (int i = threadIdx.x; i < valid; i += 256) m = fmaxf(m, p[i]);
    red[threadIdx.x] = m; __syncthreads();
    for (int st = 128; st > 0; st >>= 1) {
        if (threadIdx.x < st) red[threadIdx.x] = fmaxf(red[threadIdx.x], red[threadIdx.x + st]);
        __syncthreads();
    }
    m = red[0]; __syncthreads();

    float s = 0.f;
    for (int i = threadIdx.x; i < valid; i += 256) s += expf(p[i] - m);
    red[threadIdx.x] = s; __syncthreads();
    for (int st = 128; st > 0; st >>= 1) {
        if (threadIdx.x < st) red[threadIdx.x] += red[threadIdx.x + st];
        __syncthreads();
    }
    const float inv = 0.125f / red[0];   // softmax then / sqrt(64)
    for (int i = threadIdx.x; i < TSEQ; i += 256)
        o[i] = (i < valid) ? f2bf(expf(p[i] - m) * inv) : (us16)0;
}

// ---------------------------------------------------------------------------
// Per-row cross-entropy and mean
// ---------------------------------------------------------------------------
__global__ __launch_bounds__(256) void loss_row_kernel(
    const float* __restrict__ logits, const int* __restrict__ targets,
    float* __restrict__ rowloss)
{
    __shared__ float red[256];
    const int row = blockIdx.x;
    const float* p = logits + (long long)row * VOCAB;

    float m = -INFINITY;
    for (int i = threadIdx.x; i < VOCAB; i += 256) m = fmaxf(m, p[i]);
    red[threadIdx.x] = m; __syncthreads();
    for (int st = 128; st > 0; st >>= 1) {
        if (threadIdx.x < st) red[threadIdx.x] = fmaxf(red[threadIdx.x], red[threadIdx.x + st]);
        __syncthreads();
    }
    m = red[0]; __syncthreads();

    float s = 0.f;
    for (int i = threadIdx.x; i < VOCAB; i += 256) s += expf(p[i] - m);
    red[threadIdx.x] = s; __syncthreads();
    for (int st = 128; st > 0; st >>= 1) {
        if (threadIdx.x < st) red[threadIdx.x] += red[threadIdx.x + st];
        __syncthreads();
    }
    if (threadIdx.x == 0)
        rowloss[row] = -(p[targets[row]] - m - logf(red[0]));
}

__global__ __launch_bounds__(256) void loss_mean_kernel(
    const float* __restrict__ rowloss, float* __restrict__ out)
{
    __shared__ float red[256];
    float s = 0.f;
    for (int i = threadIdx.x; i < BBATCH * TSEQ; i += 256) s += rowloss[i];
    red[threadIdx.x] = s; __syncthreads();
    for (int st = 128; st > 0; st >>= 1) {
        if (threadIdx.x < st) red[threadIdx.x] += red[threadIdx.x + st];
        __syncthreads();
    }
    if (threadIdx.x == 0) out[0] = red[0] / (float)(BBATCH * TSEQ);
}

// ---------------------------------------------------------------------------
extern "C" void kernel_launch(void* const* d_in, const int* in_sizes, int n_in,
                              void* d_out, int out_size, void* d_ws, size_t ws_size,
                              hipStream_t stream)
{
    (void)in_sizes; (void)n_in; (void)ws_size;

    const int*   idx     = (const int*)  d_in[0];
    const int*   targets = (const int*)  d_in[1];
    const float* wte     = (const float*)d_in[2];
    const float* wpe     = (const float*)d_in[3];
    const float* ln1_s   = (const float*)d_in[4];
    const float* ln1_b   = (const float*)d_in[5];
    const float* ln2_s   = (const float*)d_in[6];
    const float* ln2_b   = (const float*)d_in[7];
    const float* Wq      = (const float*)d_in[8];
    const float* bq      = (const float*)d_in[9];
    const float* Wk      = (const float*)d_in[10];
    const float* bk      = (const float*)d_in[11];
    const float* Wv      = (const float*)d_in[12];
    const float* bv      = (const float*)d_in[13];
    const float* Wo      = (const float*)d_in[14];
    const float* bo      = (const float*)d_in[15];
    const float* W1      = (const float*)d_in[16];
    const float* b1      = (const float*)d_in[17];
    const float* W2      = (const float*)d_in[18];
    const float* b2      = (const float*)d_in[19];
    const float* lnf_s   = (const float*)d_in[20];
    const float* lnf_b   = (const float*)d_in[21];

    const long long XD   = (long long)BBATCH * TSEQ * DMODEL;
    const long long SCN  = (long long)BBATCH * NHEAD * TSEQ * TSEQ;
    const long long MLPN = (long long)BBATCH * TSEQ * FFDIM;
    const long long WQN  = (long long)LCOUNT * NHEAD * DMODEL * HSZ;
    const long long WON  = (long long)LCOUNT * DMODEL * DMODEL;
    const long long W1N  = (long long)LCOUNT * DMODEL * FFDIM;
    const long long VDN  = (long long)VOCAB * DMODEL;

    float* x       = (float*)d_ws;
    float* scores  = x + XD;
    float* rowloss = scores + SCN;
    us16*  h     = (us16*)(rowloss + 2048);
    us16*  q     = h + XD;
    us16*  k2    = q + XD;
    us16*  vT    = k2 + XD;          // [B,H,HS,T]
    us16*  att   = vT + XD;          // [B,T,D]
    us16*  probs = att + XD;         // [B,H,T,T]
    us16*  mlph  = probs + SCN;      // [B*T, F]
    us16*  wqT   = mlph + MLPN;      // [L*H, HS, D]
    us16*  wkT   = wqT + WQN;
    us16*  wvT   = wkT + WQN;
    us16*  woT   = wvT + WQN;        // [L, D, D] (transposed)
    us16*  w1T   = woT + WON;        // [L, F, D]
    us16*  w2T   = w1T + W1N;        // [L, D, F]
    us16*  wteb  = w2T + W1N;        // [V, D] bf16

    float* logits = (float*)d_out;
    float* lossp  = logits + ((long long)out_size - 1);

    const int MROWS = BBATCH * TSEQ;   // 2048
    dim3 tb(32, 8);

    // ---- one-shot weight conversion / pre-transposition (bf16, [N][K]) ----
    transpose_conv_kernel<<<dim3(2, 24, LCOUNT * NHEAD), tb, 0, stream>>>(Wq, wqT, DMODEL, HSZ);
    transpose_conv_kernel<<<dim3(2, 24, LCOUNT * NHEAD), tb, 0, stream>>>(Wk, wkT, DMODEL, HSZ);
    transpose_conv_kernel<<<dim3(2, 24, LCOUNT * NHEAD), tb, 0, stream>>>(Wv, wvT, DMODEL, HSZ);
    transpose_conv_kernel<<<dim3(24, 24, LCOUNT), tb, 0, stream>>>(Wo, woT, DMODEL, DMODEL);
    transpose_conv_kernel<<<dim3(96, 24, LCOUNT), tb, 0, stream>>>(W1, w1T, DMODEL, FFDIM);
    transpose_conv_kernel<<<dim3(24, 96, LCOUNT), tb, 0, stream>>>(W2, w2T, FFDIM, DMODEL);
    conv_bf16_kernel<<<(unsigned)((VDN / 4 + 255) / 256), 256, 0, stream>>>(wte, wteb, VDN / 4);

    // ---- embeddings ----
    embed_kernel<<<(unsigned)(XD / 256), 256, 0, stream>>>(idx, wte, wpe, x);

    for (int l = 0; l < LCOUNT; ++l) {
        const us16* wqT_l = wqT + (long long)l * NHEAD * HSZ * DMODEL;
        const us16* wkT_l = wkT + (long long)l * NHEAD * HSZ * DMODEL;
        const us16* wvT_l = wvT + (long long)l * NHEAD * HSZ * DMODEL;
        const us16* woT_l = woT + (long long)l * DMODEL * DMODEL;
        const us16* w1T_l = w1T + (long long)l * DMODEL * FFDIM;
        const us16* w2T_l = w2T + (long long)l * FFDIM * DMODEL;
        const float* bq_l = bq + (long long)l * NHEAD * HSZ;
        const float* bk_l = bk + (long long)l * NHEAD * HSZ;
        const float* bv_l = bv + (long long)l * NHEAD * HSZ;
        const float* bo_l = bo + (long long)l * DMODEL;
        const float* b1_l = b1 + (long long)l * FFDIM;
        const float* b2_l = b2 + (long long)l * DMODEL;

        ln_kernel<<<MROWS, 256, 0, stream>>>(x, ln1_s + l * DMODEL, ln1_b + l * DMODEL, h);

        // q/k: per (b,head) [T,D]x[D,HS] -> bf16 [B,H,T,HS]; v -> bf16 [B,H,HS,T]
        {
            dim3 g(1, TSEQ / 128, BBATCH * NHEAD);
            const long long sAo = (long long)TSEQ * DMODEL;
            const long long sBi = (long long)HSZ * DMODEL;
            const long long sCo = (long long)NHEAD * TSEQ * HSZ;
            const long long sCi = (long long)TSEQ * HSZ;
            gemm_bf16_wmma<GF_CBF16><<<g, 256, 0, stream>>>(h, DMODEL, sAo, 0,
                wqT_l, DMODEL, 0, sBi, q, HSZ, sCo, sCi, bq_l, HSZ,
                TSEQ, HSZ, DMODEL, NHEAD);
            gemm_bf16_wmma<GF_CBF16><<<g, 256, 0, stream>>>(h, DMODEL, sAo, 0,
                wkT_l, DMODEL, 0, sBi, k2, HSZ, sCo, sCi, bk_l, HSZ,
                TSEQ, HSZ, DMODEL, NHEAD);
            gemm_bf16_wmma<GF_CBF16 | GF_STORET><<<g, 256, 0, stream>>>(h, DMODEL, sAo, 0,
                wvT_l, DMODEL, 0, sBi, vT, TSEQ, sCo, sCi, bv_l, HSZ,
                TSEQ, HSZ, DMODEL, NHEAD);
        }

        // scores = q x k^T (k is [T,HS] == [N][K]) -> fp32
        {
            dim3 g(TSEQ / 128, TSEQ / 128, BBATCH * NHEAD);
            gemm_bf16_wmma<0><<<g, 256, 0, stream>>>(
                q, HSZ, (long long)TSEQ * HSZ, 0,
                k2, HSZ, (long long)TSEQ * HSZ, 0,
                scores, TSEQ, (long long)TSEQ * TSEQ, 0,
                nullptr, 0,
                TSEQ, TSEQ, HSZ, 1);
        }

        attn_softmax_kernel<<<BBATCH * NHEAD * TSEQ, 256, 0, stream>>>(scores, probs);

        // o = probs x v (vT is [HS][T] == [N][K]) -> bf16 att[B,T,D]
        {
            dim3 g(1, TSEQ / 128, BBATCH * NHEAD);
            gemm_bf16_wmma<GF_CBF16><<<g, 256, 0, stream>>>(
                probs, TSEQ, (long long)NHEAD * TSEQ * TSEQ, (long long)TSEQ * TSEQ,
                vT, TSEQ, (long long)NHEAD * HSZ * TSEQ, (long long)HSZ * TSEQ,
                att, DMODEL, (long long)TSEQ * DMODEL, (long long)HSZ,
                nullptr, 0,
                TSEQ, HSZ, TSEQ, NHEAD);
        }

        // x += att x Wo + bo   (WoT is [D][D])
        {
            dim3 g(DMODEL / 128, MROWS / 128, 1);
            gemm_bf16_wmma<GF_ACC><<<g, 256, 0, stream>>>(
                att, DMODEL, 0, 0, woT_l, DMODEL, 0, 0,
                x, DMODEL, 0, 0, bo_l, 0,
                MROWS, DMODEL, DMODEL, 1);
        }

        ln_kernel<<<MROWS, 256, 0, stream>>>(x, ln2_s + l * DMODEL, ln2_b + l * DMODEL, h);

        // mlph = gelu(h x W1 + b1)  (W1T is [F][D]) -> bf16
        {
            dim3 g(FFDIM / 128, MROWS / 128, 1);
            gemm_bf16_wmma<GF_GELU | GF_CBF16><<<g, 256, 0, stream>>>(
                h, DMODEL, 0, 0, w1T_l, DMODEL, 0, 0,
                mlph, FFDIM, 0, 0, b1_l, 0,
                MROWS, FFDIM, DMODEL, 1);
        }

        // x += mlph x W2 + b2  (W2T is [D][F])
        {
            dim3 g(DMODEL / 128, MROWS / 128, 1);
            gemm_bf16_wmma<GF_ACC><<<g, 256, 0, stream>>>(
                mlph, FFDIM, 0, 0, w2T_l, FFDIM, 0, 0,
                x, DMODEL, 0, 0, b2_l, 0,
                MROWS, DMODEL, FFDIM, 1);
        }
    }

    ln_kernel<<<MROWS, 256, 0, stream>>>(x, lnf_s, lnf_b, h);

    // logits = h x wte^T  (wte is [V][D] == [N][K]) -> fp32 d_out
    {
        dim3 g((VOCAB + 127) / 128, MROWS / 128, 1);
        gemm_bf16_wmma<0><<<g, 256, 0, stream>>>(
            h, DMODEL, 0, 0, wteb, DMODEL, 0, 0,
            logits, VOCAB, 0, 0, nullptr, 0,
            MROWS, VOCAB, DMODEL, 1);
    }

    loss_row_kernel<<<MROWS, 256, 0, stream>>>(logits, targets, rowloss);
    loss_mean_kernel<<<1, 256, 0, stream>>>(rowloss, lossp);
}
